// BiCRFModel_16552803959010
// MI455X (gfx1250) — matrix operand order repaired and checked by reading, not compiled, not run on hardware
//
#include <hip/hip_runtime.h>
#include <hip/hip_bf16.h>
#include <stdint.h>

// ---------------------------------------------------------------------------
// BiLSTM-CRF forward for MI455X (gfx1250), wave32 + WMMA bf16 path.
//   B=32 T=256 D=768 H=512 4H=2048 N=25
// ---------------------------------------------------------------------------

typedef __attribute__((ext_vector_type(16))) __bf16 v16bf;
typedef __attribute__((ext_vector_type(8)))  float  v8f;

#define B_    32
#define T_    256
#define D_    768
#define H_    512
#define G_    2048      // 4*H
#define N_    25
#define ROWS_ 8192      // B*T

static __device__ __forceinline__ __bf16 f2bf(float f) {
  union { float f; uint32_t u; } v; v.f = f;
  uint32_t r = v.u + 0x7fffu + ((v.u >> 16) & 1u);   // round-to-nearest-even
  uint16_t h = (uint16_t)(r >> 16);
  __bf16 b;
  __builtin_memcpy(&b, &h, 2);
  return b;
}

static __device__ __forceinline__ float sigmoidf_(float x) {
  return 1.0f / (1.0f + __expf(-x));
}

// ---------------------------------------------------------------------------
// Pack X (ROWS x D, fp32) into WMMA A-fragment layout bf16: [mt][kt][lane][16]
// A 16x32 bf16 layout (ISA 7.12.2): lane&15 = row, element j -> k =
//   ((j>>3)<<4) + ((lane>>4)<<3) + (j&7)
// ---------------------------------------------------------------------------
__global__ void pack_a_kernel(const float* __restrict__ X, __bf16* __restrict__ out) {
  int idx = blockIdx.x * blockDim.x + threadIdx.x;
  const int KT = D_ / 32;                 // 24
  int j    = idx & 15;
  int lane = (idx >> 4) & 31;
  int kt   = (idx >> 9) % KT;
  int mt   = idx / (512 * KT);
  if (mt >= ROWS_ / 16) return;
  int m = mt * 16 + (lane & 15);
  int k = kt * 32 + ((j >> 3) << 4) + ((lane >> 4) << 3) + (j & 7);
  out[idx] = f2bf(X[(size_t)m * D_ + k]);
}

// ---------------------------------------------------------------------------
// Pack W (K x 2048, fp32) into WMMA B-fragment layout bf16: [kt][nt][lane][16]
// B 32x16 bf16 layout: lane&15 = col, element j -> k =
//   ((j>>3)<<4) + ((lane>>4)<<3) + (j&7)  (matches A-fragment k order).
// ---------------------------------------------------------------------------
__global__ void pack_b_kernel(const float* __restrict__ W, __bf16* __restrict__ out, int KT) {
  int idx = blockIdx.x * blockDim.x + threadIdx.x;
  int j    = idx & 15;
  int lane = (idx >> 4) & 31;
  int nt   = (idx >> 9) & 127;            // 2048/16 = 128 n-tiles
  int kt   = idx >> 16;                   // 512*128 elems per k-tile
  if (kt >= KT) return;
  int k   = kt * 32 + ((j >> 3) << 4) + ((lane >> 4) << 3) + (j & 7);
  int col = nt * 16 + (lane & 15);
  out[idx] = f2bf(W[(size_t)k * G_ + col]);
}

// ---------------------------------------------------------------------------
// xz = X * K + bias for both directions. Block = 256 thr = 8 waves,
// wave computes 32x32 (2x2 frags); block tile 64x128. grid (16,128,2).
// ---------------------------------------------------------------------------
__global__ void gemm_xz_kernel(const __bf16* __restrict__ pA,
                               const __bf16* __restrict__ pKf,
                               const __bf16* __restrict__ pKb,
                               const float*  __restrict__ bias_f,
                               const float*  __restrict__ bias_b,
                               float* __restrict__ xzf,
                               float* __restrict__ xzb) {
  const int dir = blockIdx.z;
  const __bf16* pB  = dir ? pKb : pKf;
  const float*  bia = dir ? bias_b : bias_f;
  float*        out = dir ? xzb : xzf;

  const int l   = threadIdx.x & 31;
  const int wid = threadIdx.x >> 5;
  const int mt0 = blockIdx.y * 4 + (wid >> 2) * 2;
  const int nt0 = blockIdx.x * 8 + (wid & 3) * 2;
  const int KT  = D_ / 32;                // 24

  v8f zero8 = {0.f,0.f,0.f,0.f,0.f,0.f,0.f,0.f};
  v8f acc00 = zero8, acc01 = zero8, acc10 = zero8, acc11 = zero8;

  for (int kt = 0; kt < KT; ++kt) {
    v16bf a0 = *(const v16bf*)(pA + ((size_t)(mt0 + 0) * KT + kt) * 512 + l * 16);
    v16bf a1 = *(const v16bf*)(pA + ((size_t)(mt0 + 1) * KT + kt) * 512 + l * 16);
    v16bf b0 = *(const v16bf*)(pB + ((size_t)kt * 128 + (nt0 + 0)) * 512 + l * 16);
    v16bf b1 = *(const v16bf*)(pB + ((size_t)kt * 128 + (nt0 + 1)) * 512 + l * 16);
    acc00 = __builtin_amdgcn_wmma_f32_16x16x32_bf16(false, a0, false, b0, (short)0, acc00, false, false);
    acc01 = __builtin_amdgcn_wmma_f32_16x16x32_bf16(false, a0, false, b1, (short)0, acc01, false, false);
    acc10 = __builtin_amdgcn_wmma_f32_16x16x32_bf16(false, a1, false, b0, (short)0, acc10, false, false);
    acc11 = __builtin_amdgcn_wmma_f32_16x16x32_bf16(false, a1, false, b1, (short)0, acc11, false, false);
  }

  // C/D layout: element e, lane l -> row = e + 8*(l>>4), col = l&15
  const int rb = 8 * (l >> 4);
  const int cl = l & 15;
  v8f acc[2][2] = {{acc00, acc01}, {acc10, acc11}};
  #pragma unroll
  for (int i = 0; i < 2; ++i)
    #pragma unroll
    for (int jn = 0; jn < 2; ++jn)
      #pragma unroll
      for (int e = 0; e < 8; ++e) {
        int row = (mt0 + i) * 16 + rb + e;
        int col = (nt0 + jn) * 16 + cl;
        out[(size_t)row * G_ + col] = acc[i][jn][e] + bia[col];
      }
}

// ---------------------------------------------------------------------------
// Persistent LSTM: grid(2) = {fwd,bwd}, 1024 threads = 32 waves.
// h (32x512 bf16, A-frag layout) in LDS; c in VGPRs; rec streamed from L2.
// Wave w owns hidden columns [w*16, w*16+16) and computes all 4 gates there:
// 8 wmma accumulators (2 M-tiles x 4 gates), 16 K-steps, 128 wmma / step.
// ---------------------------------------------------------------------------
__global__ void lstm_kernel(const __bf16* __restrict__ pRf,
                            const __bf16* __restrict__ pRb,
                            const float* __restrict__ xzf,
                            const float* __restrict__ xzb,
                            float* __restrict__ hcat) {
  const int dir = blockIdx.x;
  const __bf16* recT = dir ? pRb : pRf;
  const float*  xz   = dir ? xzb : xzf;

  __shared__ __attribute__((aligned(32))) __bf16 hA[2 * 16 * 32 * 16]; // 32 KB

  const int tid = threadIdx.x;
  const int w   = tid >> 5;
  const int l   = tid & 31;

  for (int i = tid; i < 2 * 16 * 32 * 16; i += 1024) hA[i] = f2bf(0.0f);
  __syncthreads();

  float c[2][8];
  #pragma unroll
  for (int mt = 0; mt < 2; ++mt)
    #pragma unroll
    for (int e = 0; e < 8; ++e) c[mt][e] = 0.0f;

  const int col_h = w * 16 + (l & 15);       // hidden column owned
  const int rb    = 8 * (l >> 4);
  // scatter indices for writing h back into A-frag layout
  const int kt2 = col_h >> 5;
  const int k2  = col_h & 31;
  const int j2  = (k2 & 7) | ((k2 & 16) >> 1);
  const int lahi = ((k2 >> 3) & 1) << 4;

  const v8f zero8 = {0.f,0.f,0.f,0.f,0.f,0.f,0.f,0.f};

  for (int step = 0; step < T_; ++step) {
    const int t = dir ? (T_ - 1 - step) : step;

    v8f z[2][4];
    #pragma unroll
    for (int mt = 0; mt < 2; ++mt)
      #pragma unroll
      for (int g = 0; g < 4; ++g) z[mt][g] = zero8;

    for (int kt = 0; kt < 16; ++kt) {
      v16bf a0 = *(const v16bf*)(hA + ((0 * 16 + kt) * 32 + l) * 16);
      v16bf a1 = *(const v16bf*)(hA + ((1 * 16 + kt) * 32 + l) * 16);
      #pragma unroll
      for (int g = 0; g < 4; ++g) {
        const int nt = g * 32 + w;
        const __bf16* bp = recT + ((size_t)kt * 128 + nt) * 512 + l * 16;
        __builtin_prefetch(bp + 128 * 512, 0, 1);   // next k-tile -> L2/L0
        v16bf b = *(const v16bf*)bp;
        z[0][g] = __builtin_amdgcn_wmma_f32_16x16x32_bf16(false, a0, false, b, (short)0, z[0][g], false, false);
        z[1][g] = __builtin_amdgcn_wmma_f32_16x16x32_bf16(false, a1, false, b, (short)0, z[1][g], false, false);
      }
    }

    __syncthreads();   // all waves finished reading h(t-1) from LDS

    #pragma unroll
    for (int mt = 0; mt < 2; ++mt) {
      #pragma unroll
      for (int e = 0; e < 8; ++e) {
        const int b_ = mt * 16 + rb + e;                // batch row
        const size_t rbase = ((size_t)b_ * T_ + t) * (size_t)G_;
        float zi = z[mt][0][e] + xz[rbase + 0 * H_ + col_h];
        float zf = z[mt][1][e] + xz[rbase + 1 * H_ + col_h];
        float zg = z[mt][2][e] + xz[rbase + 2 * H_ + col_h];
        float zo = z[mt][3][e] + xz[rbase + 3 * H_ + col_h];
        float cn = sigmoidf_(zf) * c[mt][e] + sigmoidf_(zi) * tanhf(zg);
        float hn = sigmoidf_(zo) * tanhf(cn);
        c[mt][e] = cn;
        hcat[((size_t)b_ * T_ + t) * (size_t)(2 * H_) + dir * H_ + col_h] = hn;
        hA[(((b_ >> 4) * 16 + kt2) * 32 + ((b_ & 15) | lahi)) * 16 + j2] = f2bf(hn);
      }
    }
    __syncthreads();   // h(t) visible before next k-loop
  }
}

// ---------------------------------------------------------------------------
// logits = selu(hcat * W + b). Block = 256 (8 rows x 32 lanes), grid 1024.
// ---------------------------------------------------------------------------
__global__ void dense_selu_kernel(const float* __restrict__ hcat,
                                  const float* __restrict__ W,
                                  const float* __restrict__ bias,
                                  float* __restrict__ logits) {
  __shared__ float rows[8][2 * H_];     // 32 KB
  const int rowbase = blockIdx.x * 8;
  for (int i = threadIdx.x; i < 8 * 2 * H_; i += 256)
    rows[i >> 10][i & 1023] = hcat[((size_t)rowbase + (i >> 10)) * (2 * H_) + (i & 1023)];
  __syncthreads();

  const int r = threadIdx.x >> 5;
  const int n = threadIdx.x & 31;
  if (n < N_) {
    float s = bias[n];
    for (int k = 0; k < 2 * H_; ++k) s += rows[r][k] * W[(size_t)k * N_ + n];
    const float scale = 1.0507009873554805f, alpha = 1.6732632423543772f;
    float v = scale * (s > 0.0f ? s : alpha * (__expf(s) - 1.0f));
    logits[((size_t)(rowbase + r)) * N_ + n] = v;
  }
}

// ---------------------------------------------------------------------------
// CRF NLL: 32 warps, warp b handles batch row b; lane n holds alpha[n] (n<25).
// ---------------------------------------------------------------------------
__global__ void crf_kernel(const float* __restrict__ logits,
                           const int* __restrict__ tags,
                           const int* __restrict__ lens,
                           const float* __restrict__ trans_g,
                           float* __restrict__ out_loss) {
  __shared__ float trans[N_ * N_];
  __shared__ float nll[B_];
  const int tid = threadIdx.x;
  for (int i = tid; i < N_ * N_; i += 1024) trans[i] = trans_g[i];
  __syncthreads();

  const int b = tid >> 5;
  const int n = tid & 31;
  int L = lens[b];
  L = L < 1 ? 1 : (L > T_ ? T_ : L);

  // unary + binary scores, lane-parallel over t then shfl-reduced
  float us = 0.0f, bs = 0.0f;
  for (int t = n; t < L; t += 32)
    us += logits[((size_t)b * T_ + t) * N_ + tags[b * T_ + t]];
  for (int t = n + 1; t < L; t += 32)
    bs += trans[tags[b * T_ + t - 1] * N_ + tags[b * T_ + t]];
  for (int off = 16; off; off >>= 1) {
    us += __shfl_down(us, off, 32);
    bs += __shfl_down(bs, off, 32);
  }

  const float NEG = -1.0e30f;
  float alpha = (n < N_) ? logits[((size_t)b * T_ + 0) * N_ + n] : NEG;
  for (int t = 1; t < L; ++t) {
    float lg = (n < N_) ? logits[((size_t)b * T_ + t) * N_ + n] : 0.0f;
    float mx = NEG;
    for (int m = 0; m < N_; ++m) {
      float am = __shfl(alpha, m, 32);
      float v  = am + ((n < N_) ? trans[m * N_ + n] : NEG);
      mx = fmaxf(mx, v);
    }
    float se = 0.0f;
    for (int m = 0; m < N_; ++m) {
      float am = __shfl(alpha, m, 32);
      float v  = am + ((n < N_) ? trans[m * N_ + n] : NEG);
      se += __expf(v - mx);
    }
    float na = mx + __logf(se) + lg;
    alpha = (n < N_) ? na : NEG;
  }

  // logZ = logsumexp over n
  float mz = alpha;
  for (int off = 16; off; off >>= 1) mz = fmaxf(mz, __shfl_down(mz, off, 32));
  mz = __shfl(mz, 0, 32);
  float ez = (n < N_) ? __expf(alpha - mz) : 0.0f;
  for (int off = 16; off; off >>= 1) ez += __shfl_down(ez, off, 32);
  if (n == 0) nll[b] = -(us + bs - (mz + __logf(ez)));
  __syncthreads();
  if (tid == 0) {
    float s = 0.0f;
    for (int i = 0; i < B_; ++i) s += nll[i];
    out_loss[0] = s / (float)B_;
  }
}

// ---------------------------------------------------------------------------
extern "C" void kernel_launch(void* const* d_in, const int* in_sizes, int n_in,
                              void* d_out, int out_size, void* d_ws, size_t ws_size,
                              hipStream_t stream) {
  (void)in_sizes; (void)n_in; (void)out_size; (void)ws_size;
  const float* x     = (const float*)d_in[0];   // (32,256,768)
  const int*   tags  = (const int*)  d_in[1];   // (32,256)
  const int*   lens  = (const int*)  d_in[2];   // (32,)
  const float* kf    = (const float*)d_in[3];   // (768,2048)
  const float* rf    = (const float*)d_in[4];   // (512,2048)
  const float* bf    = (const float*)d_in[5];   // (2048,)
  const float* kb    = (const float*)d_in[6];
  const float* rb    = (const float*)d_in[7];
  const float* bb    = (const float*)d_in[8];
  const float* dw    = (const float*)d_in[9];   // (1024,25)
  const float* db    = (const float*)d_in[10];  // (25,)
  const float* trans = (const float*)d_in[11];  // (25,25)

  char* ws = (char*)d_ws;
  __bf16* pA  = (__bf16*)ws; ws += (size_t)ROWS_ * D_ * 2;       // 12.0 MB
  __bf16* pKf = (__bf16*)ws; ws += (size_t)D_ * G_ * 2;          //  3.0 MB
  __bf16* pKb = (__bf16*)ws; ws += (size_t)D_ * G_ * 2;
  __bf16* pRf = (__bf16*)ws; ws += (size_t)H_ * G_ * 2;          //  2.0 MB
  __bf16* pRb = (__bf16*)ws; ws += (size_t)H_ * G_ * 2;
  float*  xzf = (float*)ws;  ws += (size_t)ROWS_ * G_ * 4;       // 64.0 MB
  float*  xzb = (float*)ws;  ws += (size_t)ROWS_ * G_ * 4;
  float*  hct = (float*)ws;  ws += (size_t)ROWS_ * (2 * H_) * 4; // 32.0 MB

  float* out    = (float*)d_out;
  float* logits = out + 1;

  pack_a_kernel<<<(ROWS_/16) * 24 * 2, 256, 0, stream>>>(x, pA);
  pack_b_kernel<<<24 * 128 * 2, 256, 0, stream>>>(kf, pKf, 24);
  pack_b_kernel<<<24 * 128 * 2, 256, 0, stream>>>(kb, pKb, 24);
  pack_b_kernel<<<16 * 128 * 2, 256, 0, stream>>>(rf, pRf, 16);
  pack_b_kernel<<<16 * 128 * 2, 256, 0, stream>>>(rb, pRb, 16);

  gemm_xz_kernel<<<dim3(16, 128, 2), 256, 0, stream>>>(pA, pKf, pKb, bf, bb, xzf, xzb);

  lstm_kernel<<<2, 1024, 0, stream>>>(pRf, pRb, xzf, xzb, hct);

  dense_selu_kernel<<<ROWS_ / 8, 256, 0, stream>>>(hct, dw, db, logits);

  crf_kernel<<<1, 1024, 0, stream>>>(logits, tags, lens, trans, out);
}
